// FP8Linear_1340029796418
// MI455X (gfx1250) — 8-bit matrix operand order repaired and checked by reading, compiled, not run on hardware
//
#include <hip/hip_runtime.h>

#define MDIM 8192
#define KDIM 4096
#define NDIM 4096

#define BM 256
#define BN 128
#define KC 128
#define NCHUNK (KDIM / KC)

#define LANE_STRIDE   80
#define SUBTILE_BYTES (32 * LANE_STRIDE)
#define A_SUBTILES    (BM / 16)
#define B_SUBTILES    (BN / 16)
#define A_BYTES       (A_SUBTILES * SUBTILE_BYTES)
#define B_BYTES       (B_SUBTILES * SUBTILE_BYTES)
#define BUF_BYTES     (A_BYTES + B_BYTES)
#define LDS_BYTES     (2 * BUF_BYTES)

#define ASYNC_PER_WAVE 20

typedef __attribute__((ext_vector_type(16))) int   v16i;
typedef __attribute__((ext_vector_type(8)))  float v8f;

typedef int v2i __attribute__((vector_size(8)));
typedef int v4i __attribute__((vector_size(16)));

#define AS1 __attribute__((address_space(1)))
#define AS3 __attribute__((address_space(3)))

#if defined(__has_builtin)
#if __has_builtin(__builtin_amdgcn_global_load_async_to_lds_b64) && \
    __has_builtin(__builtin_amdgcn_global_load_async_to_lds_b128) && \
    __has_builtin(__builtin_amdgcn_s_wait_asynccnt)
#define HAVE_ASYNC 1
#endif
#endif
#ifndef HAVE_ASYNC
#define HAVE_ASYNC 0
#endif

__device__ __forceinline__ unsigned int quant_e4m3_byte(float f) {
    unsigned int u    = __float_as_uint(f);
    unsigned int sign = (u >> 24) & 0x80u;
    unsigned int absu = u & 0x7FFFFFFFu;
    float af          = __uint_as_float(absu);

    if (absu >= 0x7F800000u) return sign | 0x7Fu;
    if (af >= 464.0f)        return sign | 0x7Eu;
    if (af < 0.0009765625f)  return sign;
    if (af < 0.015625f) {
        int m = (int)(af * 512.0f + 0.5f);
        if (m > 7) return sign | 0x08u;
        return sign | (unsigned int)m;
    }
    unsigned int m    = absu & 0x7FFFFFu;
    unsigned int keep = m >> 20;
    unsigned int rest = m & 0xFFFFFu;
    const unsigned int half = 0x80000u;
    unsigned int mant = keep + ((rest > half || (rest == half && (keep & 1u))) ? 1u : 0u);
    int e = (int)((absu >> 23) & 0xFFu) - 127;
    if (mant == 8u) { mant = 0u; e += 1; }
    if (e > 8) return sign | 0x7Eu;
    unsigned int ee = (unsigned int)(e + 7);
    return sign | (ee << 3) | mant;
}

__global__ __launch_bounds__(256) void quantize_e4m3_kernel(
    const float* __restrict__ in, unsigned int* __restrict__ out, int n4) {
    int i = blockIdx.x * 256 + threadIdx.x;
    if (i >= n4) return;
    float4 v = ((const float4*)in)[i];
    out[i] = quant_e4m3_byte(v.x)
           | (quant_e4m3_byte(v.y) << 8)
           | (quant_e4m3_byte(v.z) << 16)
           | (quant_e4m3_byte(v.w) << 24);
}

__global__ __launch_bounds__(256) void fp8_wmma_gemm(
    const unsigned char* __restrict__ xq,
    const unsigned char* __restrict__ wq,
    float* __restrict__ out)
{
    __shared__ __align__(16) unsigned char smem[LDS_BYTES];

    const int lane = threadIdx.x & 31;
    const int wave = threadIdx.x >> 5;
    const int l16  = lane & 15;
    const int hi   = lane >> 4;

    const int wm = wave & 3;
    const int wn = wave >> 2;

    const int blockM = blockIdx.x * BM;
    const int blockN = blockIdx.y * BN;

    const unsigned char* aSrc = xq + (size_t)(blockM + 32 * wave + l16) * KDIM + 16 * hi;
    const unsigned char* bSrc = wq + (size_t)(blockN + 16 * wave + l16) * KDIM + 16 * hi;
    unsigned char* aDst = smem + (2 * wave) * SUBTILE_BYTES + (size_t)lane * LANE_STRIDE;
    unsigned char* bDst = smem + A_BYTES + wave * SUBTILE_BYTES + (size_t)lane * LANE_STRIDE;

    auto stage = [&](int buf, int k0) {
        unsigned char* ad = aDst + buf * BUF_BYTES;
        unsigned char* bd = bDst + buf * BUF_BYTES;
#pragma unroll
        for (int s = 0; s < 2; ++s) {
            const unsigned char* as_ = aSrc + (size_t)(16 * s) * KDIM + k0;
            unsigned char*       al  = ad + s * SUBTILE_BYTES;
#if HAVE_ASYNC
#pragma unroll
            for (int c = 0; c < 8; ++c)
                __builtin_amdgcn_global_load_async_to_lds_b64(
                    (AS1 v2i*)(as_ + 32 * (c >> 1) + 8 * (c & 1)),
                    (AS3 v2i*)(al + c * 8), 0, 0);
#else
#pragma unroll
            for (int c = 0; c < 8; ++c)
                *(uint2*)(al + c * 8) = *(const uint2*)(as_ + 32 * (c >> 1) + 8 * (c & 1));
#endif
        }
#if HAVE_ASYNC
#pragma unroll
        for (int c = 0; c < 4; ++c)
            __builtin_amdgcn_global_load_async_to_lds_b128(
                (AS1 v4i*)(bSrc + k0 + 32 * c),
                (AS3 v4i*)(bd + c * 16), 0, 0);
#else
#pragma unroll
        for (int c = 0; c < 4; ++c)
            *(uint4*)(bd + c * 16) = *(const uint4*)(bSrc + k0 + 32 * c);
#endif
    };

    const unsigned char* aFragBase = smem + (wm * 4) * SUBTILE_BYTES + (size_t)lane * LANE_STRIDE;
    const unsigned char* bFragBase = smem + A_BYTES + (wn * 4) * SUBTILE_BYTES + (size_t)lane * LANE_STRIDE;

    v8f acc[4][4];
#pragma unroll
    for (int i = 0; i < 4; ++i)
#pragma unroll
        for (int j = 0; j < 4; ++j)
            acc[i][j] = (v8f){};

    stage(0, 0);

    for (int it = 0; it < NCHUNK; ++it) {
        const int buf = it & 1;
        if (it + 1 < NCHUNK) {
            stage(buf ^ 1, (it + 1) * KC);
#if HAVE_ASYNC
            __builtin_amdgcn_s_wait_asynccnt(ASYNC_PER_WAVE);
#endif
        }
#if HAVE_ASYNC
        else {
            __builtin_amdgcn_s_wait_asynccnt(0);
        }
#endif
        __syncthreads();

        v16i a[4], b[4];
#pragma unroll
        for (int t = 0; t < 4; ++t) {
            const unsigned char* p = aFragBase + buf * BUF_BYTES + t * SUBTILE_BYTES;
#pragma unroll
            for (int c = 0; c < 4; ++c) {
                uint4 d = *(const uint4*)(p + 16 * c);
                a[t][4 * c + 0] = (int)d.x; a[t][4 * c + 1] = (int)d.y;
                a[t][4 * c + 2] = (int)d.z; a[t][4 * c + 3] = (int)d.w;
            }
        }
#pragma unroll
        for (int t = 0; t < 4; ++t) {
            const unsigned char* p = bFragBase + buf * BUF_BYTES + t * SUBTILE_BYTES;
#pragma unroll
            for (int c = 0; c < 4; ++c) {
                uint4 d = *(const uint4*)(p + 16 * c);
                b[t][4 * c + 0] = (int)d.x; b[t][4 * c + 1] = (int)d.y;
                b[t][4 * c + 2] = (int)d.z; b[t][4 * c + 3] = (int)d.w;
            }
        }

#pragma unroll
        for (int im = 0; im < 4; ++im)
#pragma unroll
            for (int jn = 0; jn < 4; ++jn)
                acc[im][jn] = __builtin_amdgcn_wmma_f32_16x16x128_fp8_fp8(
                    a[im], b[jn], (short)0, acc[im][jn], false, false);

        __syncthreads();
    }

    const int row0 = blockM + wm * 64 + hi * 8;
    const int col0 = blockN + wn * 64 + l16;
#pragma unroll
    for (int im = 0; im < 4; ++im)
#pragma unroll
        for (int jn = 0; jn < 4; ++jn)
#pragma unroll
            for (int v = 0; v < 8; ++v)
                out[(size_t)(row0 + 16 * im + v) * NDIM + col0 + 16 * jn] = acc[im][jn][v];
}

extern "C" void kernel_launch(void* const* d_in, const int* in_sizes, int n_in,
                              void* d_out, int out_size, void* d_ws, size_t ws_size,
                              hipStream_t stream) {
    (void)in_sizes; (void)n_in; (void)out_size; (void)ws_size;

    const float* x = (const float*)d_in[0];
    const float* w = (const float*)d_in[1];
    float* out     = (float*)d_out;

    unsigned char* xq = (unsigned char*)d_ws;
    unsigned char* wq = xq + (size_t)MDIM * KDIM;

    const int n4x = MDIM * KDIM / 4;
    const int n4w = NDIM * KDIM / 4;
    quantize_e4m3_kernel<<<n4x / 256, 256, 0, stream>>>(x, (unsigned int*)xq, n4x);
    quantize_e4m3_kernel<<<n4w / 256, 256, 0, stream>>>(w, (unsigned int*)wq, n4w);

    dim3 grid(MDIM / BM, NDIM / BN);
    fp8_wmma_gemm<<<grid, 256, 0, stream>>>(xq, wq, out);
}
